// CausalSelfAttention_36756330119179
// MI455X (gfx1250) — compile-verified, matrix operand order checked by reading, not run on hardware
//
#include <hip/hip_runtime.h>

// ---------------------------------------------------------------------------
// CausalSelfAttention for MI455X (gfx1250): bf16 WMMA pipeline, f32 accumulate
// ---------------------------------------------------------------------------
#define EMBC   1024
#define HEADSC 16
#define HDC    64
#define BC     2
#define SC     2048
#define MROWS  (BC * SC)          // 4096 token rows

typedef __attribute__((ext_vector_type(16))) __bf16 v16bf;
typedef __attribute__((ext_vector_type(8)))  __bf16 v8bf;
typedef __attribute__((ext_vector_type(8)))  float  v8f;

// fp32 -> bf16 round-to-nearest-even (pure bit ops; no __bf16 arithmetic)
static __device__ __forceinline__ __bf16 f2bf(float f) {
    unsigned int u = __builtin_bit_cast(unsigned int, f);
    u = (u + 0x7FFFu + ((u >> 16) & 1u)) >> 16;
    unsigned short s = (unsigned short)u;
    return __builtin_bit_cast(__bf16, s);
}
static __device__ __forceinline__ float bf2f(__bf16 b) {
    unsigned int u = ((unsigned int)__builtin_bit_cast(unsigned short, b)) << 16;
    return __builtin_bit_cast(float, u);
}

// Load one 16-element bf16 fragment (A or B matrix, 16x32 / 32x16 layout).
// Per the CDNA5 ISA layout, each lane holds two contiguous 8-element runs of K,
// so two 16-byte loads cover it.
static __device__ __forceinline__ v16bf load_frag(const __bf16* plo, const __bf16* phi) {
    v8bf lo = *(const v8bf*)plo;
    v8bf hi = *(const v8bf*)phi;
    v16bf r;
#pragma unroll
    for (int j = 0; j < 8; ++j) { r[j] = lo[j]; r[j + 8] = hi[j]; }
    return r;
}

static __device__ __forceinline__ v8f wmma_bf16(v16bf a, v16bf b, v8f c) {
    // (neg_a, A, neg_b, B, c_mod, C, reuse_a, reuse_b)
    return __builtin_amdgcn_wmma_f32_16x16x32_bf16(false, a, false, b, (short)0, c,
                                                   false, false);
}

// row reduction across the 16 lanes that hold one C/D-matrix row (xor<16 stays
// inside a 16-lane half of the wave32)
static __device__ __forceinline__ float rmax16(float v) {
#pragma unroll
    for (int m = 1; m < 16; m <<= 1) v = fmaxf(v, __shfl_xor(v, m, 32));
    return v;
}
static __device__ __forceinline__ float rsum16(float v) {
#pragma unroll
    for (int m = 1; m < 16; m <<= 1) v += __shfl_xor(v, m, 32);
    return v;
}

// ---------------------------------------------------------------------------
// Kernel 1: fp32 -> bf16 conversion
// ---------------------------------------------------------------------------
__global__ void k_cvt_bf16(const float* __restrict__ src, __bf16* __restrict__ dst, int n) {
    int i = blockIdx.x * blockDim.x + threadIdx.x;
    if (i < n) dst[i] = f2bf(src[i]);
}

// ---------------------------------------------------------------------------
// Kernel 2: fused Q/K/V projection GEMM.  C = x @ W.T  (M=4096, N=1024, K=1024)
// One wave = one 16x16 tile of each of Q,K,V (A fragment reused 3x).
// Q,K stored (b,h,s,d); V stored transposed (b,h,d,s) for the PV stage.
// grid(8,256) block(256) -> nt = bx*8+wave in [0,64), mt = by in [0,256)
// ---------------------------------------------------------------------------
__global__ void k_qkv_gemm(const __bf16* __restrict__ xb,
                           const __bf16* __restrict__ wq,
                           const __bf16* __restrict__ wk,
                           const __bf16* __restrict__ wv,
                           __bf16* __restrict__ qb,
                           __bf16* __restrict__ kb,
                           __bf16* __restrict__ vtb) {
    const int lane = threadIdx.x & 31;
    const int wave = threadIdx.x >> 5;
    const int half = lane >> 4;
    const int ln   = lane & 15;
    const int mt = blockIdx.y;
    const int nt = blockIdx.x * 8 + wave;

    const __bf16* ap = xb + (mt * 16 + ln) * EMBC;   // A: M = lane%16
    const __bf16* bqp = wq + (nt * 16 + ln) * EMBC;  // B: N = lane%16 (W row-major in K)
    const __bf16* bkp = wk + (nt * 16 + ln) * EMBC;
    const __bf16* bvp = wv + (nt * 16 + ln) * EMBC;

    v8f accq = {0,0,0,0,0,0,0,0}, acck = accq, accv = accq;

    for (int kk = 0; kk < EMBC; kk += 32) {
        const int off = kk + half * 8;
        v16bf A  = load_frag(ap  + off, ap  + off + 16);
        v16bf Bq = load_frag(bqp + off, bqp + off + 16);
        v16bf Bk = load_frag(bkp + off, bkp + off + 16);
        v16bf Bv = load_frag(bvp + off, bvp + off + 16);
        accq = wmma_bf16(A, Bq, accq);
        acck = wmma_bf16(A, Bk, acck);
        accv = wmma_bf16(A, Bv, accv);
    }

    const int col = nt * 16 + ln;         // e-dimension
    const int h   = col / HDC;
    const int d   = col % HDC;
#pragma unroll
    for (int i = 0; i < 8; ++i) {
        const int m = mt * 16 + i + 8 * half;   // token row
        const int b = m / SC;
        const int s = m % SC;
        const long bh = (long)b * HEADSC + h;
        qb[(bh * SC + s) * HDC + d]  = f2bf(accq[i]);
        kb[(bh * SC + s) * HDC + d]  = f2bf(acck[i]);
        vtb[(bh * HDC + d) * SC + s] = f2bf(accv[i]);   // transposed V
    }
}

// ---------------------------------------------------------------------------
// Kernel 3: RoPE on Q and K (pair per thread, matches reference's
// concat([freqs,freqs]) angles with interleaved rotate_half)
// ---------------------------------------------------------------------------
__global__ void k_rope(__bf16* __restrict__ qb, __bf16* __restrict__ kb) {
    const int idx = blockIdx.x * blockDim.x + threadIdx.x;   // [0, 2*2^21)
    const int t   = idx >> 21;                // 0=Q, 1=K
    const int r   = idx & ((1 << 21) - 1);    // pair index within tensor
    const int p   = r & 31;                   // pair id in head_dim (d0 = 2p)
    const int row = r >> 5;                   // (b,h,s) flat
    const int s   = row & (SC - 1);
    __bf16* buf = t ? kb : qb;
    const long base = (long)row * HDC + 2 * p;

    const float x0 = bf2f(buf[base]);
    const float x1 = bf2f(buf[base + 1]);
    const float c  = 0.28782313662f;          // ln(10000)/32
    const float a0 = (float)s * __expf(-(float)((2 * p)     & 31) * c);
    const float a1 = (float)s * __expf(-(float)((2 * p + 1) & 31) * c);
    buf[base]     = f2bf(x0 * __cosf(a0) - x1 * __sinf(a0));
    buf[base + 1] = f2bf(x1 * __cosf(a1) + x0 * __sinf(a1));
}

// ---------------------------------------------------------------------------
// Kernel 4: flash attention.  One wave per 16-query tile; 32-key blocks.
// scores: 4x wmma (K=64).  P re-laid out via per-wave LDS.  PV: 4x wmma.
// grid(16, 32) block(256): qtile = bx*8+wave in [0,128), bh = by in [0,32)
// ---------------------------------------------------------------------------
__global__ void k_flash(const __bf16* __restrict__ qb,
                        const __bf16* __restrict__ kb,
                        const __bf16* __restrict__ vtb,
                        __bf16* __restrict__ attnb) {
    __shared__ __align__(16) __bf16 pbuf[8][16 * 32];

    const int lane = threadIdx.x & 31;
    const int wave = threadIdx.x >> 5;
    const int half = lane >> 4;
    const int ln   = lane & 15;
    const int bh = blockIdx.y;
    const int q0 = (blockIdx.x * 8 + wave) * 16;

    // Q as A-matrix (16x64): lane row = q0 + lane%16, two K-chunks
    const __bf16* qr = qb + ((long)bh * SC + q0 + ln) * HDC;
    const v16bf Aq0 = load_frag(qr + half * 8,      qr + 16 + half * 8);
    const v16bf Aq1 = load_frag(qr + 32 + half * 8, qr + 48 + half * 8);

    v8f o[4];
    float mrow[8], lrow[8];
#pragma unroll
    for (int i = 0; i < 8; ++i) { mrow[i] = -1e30f; lrow[i] = 0.0f; }
#pragma unroll
    for (int ntv = 0; ntv < 4; ++ntv) o[ntv] = (v8f){0,0,0,0,0,0,0,0};

    const float scale = 0.125f;   // 1/sqrt(64)
    const int nkb = q0 / 32 + 1;  // causal: keys [0, q0+15]

    for (int kbi = 0; kbi < nkb; ++kbi) {
        const int kb0 = kbi * 32;
        // ---- scores = Q @ K^T for 32 keys (two 16-key subtiles) ----
        const __bf16* kr0 = kb + ((long)bh * SC + kb0 + ln) * HDC;       // keys kb0..+15
        const __bf16* kr1 = kr0 + 16 * HDC;                              // keys kb0+16..+31
        v8f s0 = {0,0,0,0,0,0,0,0}, s1 = s0;
        s0 = wmma_bf16(Aq0, load_frag(kr0 +      half * 8, kr0 + 16 + half * 8), s0);
        s0 = wmma_bf16(Aq1, load_frag(kr0 + 32 + half * 8, kr0 + 48 + half * 8), s0);
        s1 = wmma_bf16(Aq0, load_frag(kr1 +      half * 8, kr1 + 16 + half * 8), s1);
        s1 = wmma_bf16(Aq1, load_frag(kr1 + 32 + half * 8, kr1 + 48 + half * 8), s1);

        const bool needmask = (kb0 + 31 > q0);
        // ---- online softmax update (row M = i + 8*half, col N = lane%16) ----
#pragma unroll
        for (int i = 0; i < 8; ++i) {
            const int qg = q0 + i + 8 * half;
            float v0 = s0[i] * scale;
            float v1 = s1[i] * scale;
            if (needmask) {
                if (kb0 + ln      > qg) v0 = -1e30f;
                if (kb0 + 16 + ln > qg) v1 = -1e30f;
            }
            const float rm = rmax16(fmaxf(v0, v1));
            const float mn = fmaxf(mrow[i], rm);
            const float al = __expf(mrow[i] - mn);
            const float p0 = __expf(v0 - mn);
            const float p1 = __expf(v1 - mn);
            const float rs = rsum16(p0 + p1);
            lrow[i] = lrow[i] * al + rs;
            mrow[i] = mn;
            o[0][i] *= al; o[1][i] *= al; o[2][i] *= al; o[3][i] *= al;
            pbuf[wave][(i + 8 * half) * 32 + ln]      = f2bf(p0);
            pbuf[wave][(i + 8 * half) * 32 + 16 + ln] = f2bf(p1);
        }
        // in-wave LDS RAW: DS ops are in-order per wave; make it explicit
        asm volatile("s_wait_dscnt 0" ::: "memory");

        // ---- re-read P as A-matrix (16x32) fragment ----
        const __bf16* prow = &pbuf[wave][ln * 32 + half * 8];
        const v16bf Ap = load_frag(prow, prow + 16);

        // ---- out += P @ V  (V transposed: rows = d, contiguous keys) ----
#pragma unroll
        for (int ntv = 0; ntv < 4; ++ntv) {
            const __bf16* vr = vtb + ((long)bh * HDC + ntv * 16 + ln) * SC + kb0;
            const v16bf Bv = load_frag(vr + half * 8, vr + 16 + half * 8);
            o[ntv] = wmma_bf16(Ap, Bv, o[ntv]);
        }
    }

    // ---- normalize and store to attn (b, s, e) bf16 ----
    const int b = bh >> 4;
    const int h = bh & 15;
#pragma unroll
    for (int i = 0; i < 8; ++i) {
        const float inv = 1.0f / lrow[i];
        const int sg = q0 + i + 8 * half;
        __bf16* dst = attnb + ((long)b * SC + sg) * EMBC + h * HDC + ln;
#pragma unroll
        for (int ntv = 0; ntv < 4; ++ntv)
            dst[ntv * 16] = f2bf(o[ntv][i] * inv);
    }
}

// ---------------------------------------------------------------------------
// Kernel 5: output projection.  out = attn @ Wo.T -> fp32
// grid(8,256) block(256)
// ---------------------------------------------------------------------------
__global__ void k_outproj(const __bf16* __restrict__ attnb,
                          const __bf16* __restrict__ wo,
                          float* __restrict__ out) {
    const int lane = threadIdx.x & 31;
    const int wave = threadIdx.x >> 5;
    const int half = lane >> 4;
    const int ln   = lane & 15;
    const int mt = blockIdx.y;
    const int nt = blockIdx.x * 8 + wave;

    const __bf16* ap = attnb + (mt * 16 + ln) * EMBC;
    const __bf16* bp = wo    + (nt * 16 + ln) * EMBC;

    v8f acc = {0,0,0,0,0,0,0,0};
    for (int kk = 0; kk < EMBC; kk += 32) {
        const int off = kk + half * 8;
        v16bf A = load_frag(ap + off, ap + off + 16);
        v16bf B = load_frag(bp + off, bp + off + 16);
        acc = wmma_bf16(A, B, acc);
    }
#pragma unroll
    for (int i = 0; i < 8; ++i) {
        const int m = mt * 16 + i + 8 * half;
        out[(long)m * EMBC + nt * 16 + ln] = acc[i];
    }
}

// ---------------------------------------------------------------------------
// Host launcher
// ---------------------------------------------------------------------------
extern "C" void kernel_launch(void* const* d_in, const int* in_sizes, int n_in,
                              void* d_out, int out_size, void* d_ws, size_t ws_size,
                              hipStream_t stream) {
    (void)in_sizes; (void)n_in; (void)out_size; (void)ws_size;
    const float* x  = (const float*)d_in[0];
    const float* Wq = (const float*)d_in[1];
    const float* Wk = (const float*)d_in[2];
    const float* Wv = (const float*)d_in[3];
    const float* Wo = (const float*)d_in[4];
    float* out = (float*)d_out;

    // workspace layout (bytes), total 48 MB
    char* ws = (char*)d_ws;
    __bf16* xb  = (__bf16*)(ws + 0);                          // 8 MB
    __bf16* wqb = (__bf16*)(ws + (8u  << 20));                // 2 MB
    __bf16* wkb = (__bf16*)(ws + (10u << 20));
    __bf16* wvb = (__bf16*)(ws + (12u << 20));
    __bf16* wob = (__bf16*)(ws + (14u << 20));
    __bf16* qb  = (__bf16*)(ws + (16u << 20));                // 8 MB
    __bf16* kb  = (__bf16*)(ws + (24u << 20));                // 8 MB
    __bf16* vtb = (__bf16*)(ws + (32u << 20));                // 8 MB (b,h,d,s)
    __bf16* atn = (__bf16*)(ws + (40u << 20));                // 8 MB (b,s,e)

    const int nX = MROWS * EMBC;      // 4194304
    const int nW = EMBC * EMBC;       // 1048576
    k_cvt_bf16<<<nX / 256, 256, 0, stream>>>(x,  xb,  nX);
    k_cvt_bf16<<<nW / 256, 256, 0, stream>>>(Wq, wqb, nW);
    k_cvt_bf16<<<nW / 256, 256, 0, stream>>>(Wk, wkb, nW);
    k_cvt_bf16<<<nW / 256, 256, 0, stream>>>(Wv, wvb, nW);
    k_cvt_bf16<<<nW / 256, 256, 0, stream>>>(Wo, wob, nW);

    k_qkv_gemm<<<dim3(8, MROWS / 16), 256, 0, stream>>>(xb, wqb, wkb, wvb, qb, kb, vtb);

    const int ropeThreads = 2 * (BC * HEADSC * SC) * (HDC / 2);   // 4194304
    k_rope<<<ropeThreads / 256, 256, 0, stream>>>(qb, kb);

    k_flash<<<dim3((SC / 16) / 8, BC * HEADSC), 256, 0, stream>>>(qb, kb, vtb, atn);

    k_outproj<<<dim3(8, MROWS / 16), 256, 0, stream>>>(atn, wob, out);
}
